// Seq2Seq_20555713478733
// MI455X (gfx1250) — compile-verified, matrix operand order checked by reading, not run on hardware
//
#include <hip/hip_runtime.h>
#include <math.h>

#define BB 128
#define SS 120
#define TT 120
#define HE 256     // H_ENC
#define HD 512     // H_DEC
#define EE 5       // E_ENC
#define ED 128     // E_DEC
#define VC 100     // V_CMP

typedef __attribute__((ext_vector_type(16))) _Float16 v16h;
typedef __attribute__((ext_vector_type(8)))  _Float16 v8h;
typedef __attribute__((ext_vector_type(8)))  float    v8f;

__device__ __forceinline__ float sigf(float x) { return 1.f / (1.f + expf(-x)); }

// ---- WMMA fragment helper (CDNA5 wave32, v_wmma_f32_16x16x32_f16) ----
// 16x32 f16 fragment: lane&15 = row, two 8-half chunks at K offsets
// (lane>>4)*8 and 16+(lane>>4)*8. Same addressing serves A (row-major
// activations) and B (row-major NxK weights, i.e. W^T in the product).
__device__ __forceinline__ v16h frag_f16(const _Float16* __restrict__ tile, int ld, int lane) {
    int r = lane & 15;
    int ko = (lane >> 4) << 3;
    const _Float16* p = tile + r * ld + ko;
    v8h lo = *(const v8h*)p;
    v8h hi = *(const v8h*)(p + 16);
    v16h f;
#pragma unroll
    for (int i = 0; i < 8; ++i) { f[i] = lo[i]; f[i + 8] = hi[i]; }
    return f;
}
__device__ __forceinline__ v8f wmma32(v16h a, v16h b, v8f c) {
    return __builtin_amdgcn_wmma_f32_16x16x32_f16(false, a, false, b, (short)0, c, false, false);
}

// ---- f32 -> f16 conversion (optional column slice / row zero-pad) ----
__global__ void cvt_kernel(const float* __restrict__ src, _Float16* __restrict__ dst,
                           int dst_rows, int dst_cols, int src_rows, int src_cols, int col_off) {
    int idx = blockIdx.x * blockDim.x + threadIdx.x;
    if (idx >= dst_rows * dst_cols) return;
    int r = idx / dst_cols, c = idx % dst_cols;
    float v = (r < src_rows) ? src[r * src_cols + col_off + c] : 0.f;
    dst[idx] = (_Float16)v;
}

// zero an f32 buffer and its f16 mirror
__global__ void zero_pair_kernel(float* __restrict__ p, _Float16* __restrict__ ph, int n) {
    int i = blockIdx.x * blockDim.x + threadIdx.x;
    if (i < n) { p[i] = 0.f; ph[i] = (_Float16)0.f; }
}

__global__ void h0_init_kernel(const float* __restrict__ hf, const float* __restrict__ hb,
                               float* __restrict__ h0, _Float16* __restrict__ h0h) {
    int idx = blockIdx.x * blockDim.x + threadIdx.x;
    if (idx >= BB * HD) return;
    int b = idx / HD, j = idx % HD;
    float v = (j < HE) ? hf[b * HE + j] : hb[b * HE + (j - HE)];
    h0[idx] = v;
    h0h[idx] = (_Float16)v;
}

// ---- GEMM: C(MxN,f32) = A(MxK,f16) @ W(NxK,f16)^T (+bias), 8 waves/block ----
__global__ void gemm_kernel(const _Float16* __restrict__ A, const _Float16* __restrict__ W,
                            const float* __restrict__ bias, float* __restrict__ C,
                            int K, int N) {
    int wave = threadIdx.x >> 5, lane = threadIdx.x & 31;
    int rt = blockIdx.x;
    int ct = blockIdx.y * 8 + wave;
    const _Float16* Ab = A + (size_t)rt * 16 * K;
    const _Float16* Wb = W + (size_t)ct * 16 * K;
    v8f acc = {};
    for (int kc = 0; kc < K; kc += 32)
        acc = wmma32(frag_f16(Ab + kc, K, lane), frag_f16(Wb + kc, K, lane), acc);
    int c = ct * 16 + (lane & 15);
    int r0 = rt * 16 + ((lane >> 4) << 3);
    float bv = bias ? bias[c] : 0.f;
#pragma unroll
    for (int v = 0; v < 8; ++v) C[(size_t)(r0 + v) * N + c] = acc[v] + bv;
}

// ---- Fused encoder GRU step (both directions); gh via WMMA, gi (E=5) scalar ----
// grid (8 strips, 2 dirs), block 512 = 16 waves; wave w owns gate-col tile w of 16.
__global__ void enc_step_kernel(const int* __restrict__ tgts, const float* __restrict__ emb_enc,
                                const float* __restrict__ Wih_f, const float* __restrict__ bih_f,
                                const float* __restrict__ bhh_f, const _Float16* __restrict__ Whh16_f,
                                float* __restrict__ h_f, _Float16* __restrict__ h16_f,
                                const float* __restrict__ Wih_b, const float* __restrict__ bih_b,
                                const float* __restrict__ bhh_b, const _Float16* __restrict__ Whh16_b,
                                float* __restrict__ h_b, _Float16* __restrict__ h16_b,
                                float* __restrict__ enc_out, _Float16* __restrict__ enc_outh, int t) {
    int dir = blockIdx.y;
    int tt = dir ? (SS - 1 - t) : t;
    const float* Wih = dir ? Wih_b : Wih_f;
    const float* bih = dir ? bih_b : bih_f;
    const float* bhh = dir ? bhh_b : bhh_f;
    const _Float16* Whh = dir ? Whh16_b : Whh16_f;
    float* h = dir ? h_b : h_f;
    _Float16* h16 = dir ? h16_b : h16_f;
    int strip = blockIdx.x;
    int wave = threadIdx.x >> 5, lane = threadIdx.x & 31;
    const _Float16* Ab = h16 + strip * 16 * HE;
    const _Float16* Wr = Whh + (wave * 16) * HE;
    const _Float16* Wz = Whh + (HE + wave * 16) * HE;
    const _Float16* Wn = Whh + (2 * HE + wave * 16) * HE;
    v8f ar = {}, az = {}, an = {};
    for (int kc = 0; kc < HE; kc += 32) {
        v16h a = frag_f16(Ab + kc, HE, lane);
        ar = wmma32(a, frag_f16(Wr + kc, HE, lane), ar);
        az = wmma32(a, frag_f16(Wz + kc, HE, lane), az);
        an = wmma32(a, frag_f16(Wn + kc, HE, lane), an);
    }
    __syncthreads();   // all waves finished reading h16 strip before in-place update
    int c = wave * 16 + (lane & 15);
    int rb = (lane >> 4) << 3;
    float bir = bih[c], biz = bih[HE + c], bin = bih[2 * HE + c];
    float bhr = bhh[c], bhz = bhh[HE + c], bhn = bhh[2 * HE + c];
    const float* wir = Wih + c * EE;
    const float* wiz = Wih + (HE + c) * EE;
    const float* win = Wih + (2 * HE + c) * EE;
#pragma unroll
    for (int v = 0; v < 8; ++v) {
        int row = strip * 16 + rb + v;
        int tok = tgts[row * SS + tt];
        const float* xe = emb_enc + tok * EE;
        float gir = bir, giz = biz, gin = bin;
#pragma unroll
        for (int e = 0; e < EE; ++e) {
            float xv = xe[e];
            gir += xv * wir[e]; giz += xv * wiz[e]; gin += xv * win[e];
        }
        float r = sigf(gir + ar[v] + bhr);
        float z = sigf(giz + az[v] + bhz);
        float nn = tanhf(gin + r * (an[v] + bhn));
        float hp = h[row * HE + c];
        float hnew = (1.f - z) * nn + z * hp;
        h[row * HE + c] = hnew;
        h16[row * HE + c] = (_Float16)hnew;
        size_t eo = ((size_t)row * SS + tt) * (2 * HE) + dir * HE + c;
        enc_out[eo] = hnew;
        enc_outh[eo] = (_Float16)hnew;
    }
}

// ---- Fused decoder GRU layer: gi,gh via WMMA, cell in registers, in-place h ----
// grid 8 strips, block 512 = 16 waves; wave w owns gate-col tiles 2w,2w+1 (of 32)
// for all three gates -> r/z/n accumulators are element-aligned.
__global__ void dec_gru_kernel(const _Float16* __restrict__ x16, int inK,
                               const _Float16* __restrict__ Wih, const _Float16* __restrict__ Whh,
                               const float* __restrict__ bih, const float* __restrict__ bhh,
                               float* __restrict__ h, _Float16* __restrict__ h16) {
    int strip = blockIdx.x;
    int wave = threadIdx.x >> 5, lane = threadIdx.x & 31;
    const _Float16* Ax = x16 + (size_t)strip * 16 * inK;
    const _Float16* Ah = h16 + strip * 16 * HD;
    v8f gi[6], gh[6];
#pragma unroll
    for (int i = 0; i < 6; ++i) { gi[i] = (v8f){}; gh[i] = (v8f){}; }
    for (int kc = 0; kc < inK; kc += 32) {
        v16h a = frag_f16(Ax + kc, inK, lane);
#pragma unroll
        for (int g = 0; g < 3; ++g)
#pragma unroll
            for (int j = 0; j < 2; ++j) {
                const _Float16* wb = Wih + (size_t)(g * HD + (2 * wave + j) * 16) * inK + kc;
                gi[g * 2 + j] = wmma32(a, frag_f16(wb, inK, lane), gi[g * 2 + j]);
            }
    }
    for (int kc = 0; kc < HD; kc += 32) {
        v16h a = frag_f16(Ah + kc, HD, lane);
#pragma unroll
        for (int g = 0; g < 3; ++g)
#pragma unroll
            for (int j = 0; j < 2; ++j) {
                const _Float16* wb = Whh + (size_t)(g * HD + (2 * wave + j) * 16) * HD + kc;
                gh[g * 2 + j] = wmma32(a, frag_f16(wb, HD, lane), gh[g * 2 + j]);
            }
    }
    __syncthreads();   // finish all reads of h16 strip before in-place write
    int n = lane & 15, rb = (lane >> 4) << 3;
#pragma unroll
    for (int j = 0; j < 2; ++j) {
        int c = (2 * wave + j) * 16 + n;
        float bir = bih[c], bhr = bhh[c];
        float biz = bih[HD + c], bhz = bhh[HD + c];
        float bin = bih[2 * HD + c], bhn = bhh[2 * HD + c];
#pragma unroll
        for (int v = 0; v < 8; ++v) {
            int row = strip * 16 + rb + v;
            float r = sigf(gi[0 + j][v] + bir + gh[0 + j][v] + bhr);
            float z = sigf(gi[2 + j][v] + biz + gh[2 + j][v] + bhz);
            float nn = tanhf(gi[4 + j][v] + bin + r * (gh[4 + j][v] + bhn));
            float hp = h[row * HD + c];
            float hnew = (1.f - z) * nn + z * hp;
            h[row * HD + c] = hnew;
            h16[row * HD + c] = (_Float16)hnew;
        }
    }
}

// ---- Attention: score=tanh(q+pre)@v, softmax over S, ctx, build inp16=[emb,ctx] ----
__global__ void attn_ctx_kernel(const float* __restrict__ q, const float* __restrict__ pre,
                                const float* __restrict__ attn_v, const float* __restrict__ enc_out,
                                const float* __restrict__ emb_dec, const int* __restrict__ cmps,
                                _Float16* __restrict__ inp16, int t) {
    __shared__ float sv[HD], sq[HD], sa[SS], red[128];
    int b = blockIdx.x, tid = threadIdx.x;
    for (int i = tid; i < HD; i += 128) { sv[i] = attn_v[i]; sq[i] = q[b * HD + i]; }
    __syncthreads();
    float sc = -1e30f;
    if (tid < SS) {
        const float* pr = pre + (size_t)(b * SS + tid) * HD;
        float a = 0.f;
        for (int hh = 0; hh < HD; ++hh) a += sv[hh] * tanhf(sq[hh] + pr[hh]);
        sc = a;
    }
    red[tid] = sc;
    __syncthreads();
    for (int s = 64; s > 0; s >>= 1) { if (tid < s) red[tid] = fmaxf(red[tid], red[tid + s]); __syncthreads(); }
    float mx = red[0];
    __syncthreads();
    float e = (tid < SS) ? expf(sc - mx) : 0.f;
    red[tid] = e;
    __syncthreads();
    for (int s = 64; s > 0; s >>= 1) { if (tid < s) red[tid] += red[tid + s]; __syncthreads(); }
    float inv = 1.f / red[0];
    if (tid < SS) sa[tid] = e * inv;
    __syncthreads();
    for (int hh = tid; hh < HD; hh += 128) {
        float c = 0.f;
        const float* eo = enc_out + (size_t)b * SS * HD + hh;
        for (int s = 0; s < SS; ++s) c += sa[s] * eo[(size_t)s * HD];
        inp16[b * (ED + HD) + ED + hh] = (_Float16)c;
    }
    int tok = (t == 0) ? 0 : cmps[b * TT + (t - 1)];
    if (tid < ED) inp16[b * (ED + HD) + tid] = (_Float16)emb_dec[tok * ED + tid];
}

// ---- Logits (WMMA, out_W zero-padded to 112 rows) + log_softmax over 100 ----
__global__ void logits_kernel(const _Float16* __restrict__ h2h, const _Float16* __restrict__ Wpad,
                              const float* __restrict__ out_b, float* __restrict__ out, int t) {
    __shared__ float sl[16][112];
    int strip = blockIdx.x;
    int wave = threadIdx.x >> 5, lane = threadIdx.x & 31;
    if (wave < 7) {
        const _Float16* Ab = h2h + strip * 16 * HD;
        const _Float16* Wb = Wpad + (size_t)wave * 16 * HD;
        v8f acc = {};
        for (int kc = 0; kc < HD; kc += 32)
            acc = wmma32(frag_f16(Ab + kc, HD, lane), frag_f16(Wb + kc, HD, lane), acc);
        int c = wave * 16 + (lane & 15);
        int rb = (lane >> 4) << 3;
        if (c < VC) {
#pragma unroll
            for (int v = 0; v < 8; ++v) sl[rb + v][c] = acc[v] + out_b[c];
        }
    }
    __syncthreads();
    if (threadIdx.x < 16) {
        int r = threadIdx.x;
        int gb = strip * 16 + r;
        float mx = -1e30f;
        for (int v = 0; v < VC; ++v) mx = fmaxf(mx, sl[r][v]);
        float s = 0.f;
        for (int v = 0; v < VC; ++v) s += expf(sl[r][v] - mx);
        float lse = mx + logf(s);
        float* o = out + ((size_t)gb * TT + t) * VC;
        for (int v = 0; v < VC; ++v) o[v] = sl[r][v] - lse;
    }
}

extern "C" void kernel_launch(void* const* d_in, const int* in_sizes, int n_in,
                              void* d_out, int out_size, void* d_ws, size_t ws_size,
                              hipStream_t stream) {
    const int* tgts = (const int*)d_in[0];
    const int* cmps = (const int*)d_in[1];
    const float* emb_enc = (const float*)d_in[2];
    const float* eWih_f = (const float*)d_in[3];
    const float* eWhh_f = (const float*)d_in[4];
    const float* ebih_f = (const float*)d_in[5];
    const float* ebhh_f = (const float*)d_in[6];
    const float* eWih_b = (const float*)d_in[7];
    const float* eWhh_b = (const float*)d_in[8];
    const float* ebih_b = (const float*)d_in[9];
    const float* ebhh_b = (const float*)d_in[10];
    const float* emb_dec = (const float*)d_in[11];
    const float* attn_W = (const float*)d_in[12];
    const float* attn_b = (const float*)d_in[13];
    const float* attn_v = (const float*)d_in[14];
    const float* dWih[3] = {(const float*)d_in[15], (const float*)d_in[19], (const float*)d_in[23]};
    const float* dWhh[3] = {(const float*)d_in[16], (const float*)d_in[20], (const float*)d_in[24]};
    const float* dbih[3] = {(const float*)d_in[17], (const float*)d_in[21], (const float*)d_in[25]};
    const float* dbhh[3] = {(const float*)d_in[18], (const float*)d_in[22], (const float*)d_in[26]};
    const float* out_W = (const float*)d_in[27];
    const float* out_b = (const float*)d_in[28];
    float* out = (float*)d_out;

    // ---- workspace bump allocator (256B aligned) ----
    char* ws = (char*)d_ws;
    size_t off = 0;
    auto alloc = [&](size_t bytes) -> char* {
        char* p = ws + off;
        off += (bytes + 255) & ~(size_t)255;
        return p;
    };
    float* enc_out = (float*)alloc(sizeof(float) * BB * SS * 2 * HE);
    float* pre_enc = (float*)alloc(sizeof(float) * BB * SS * HD);
    float* h_f = (float*)alloc(sizeof(float) * BB * HE);
    float* h_b = (float*)alloc(sizeof(float) * BB * HE);
    float* h0  = (float*)alloc(sizeof(float) * BB * HD);
    float* h1  = (float*)alloc(sizeof(float) * BB * HD);
    float* h2  = (float*)alloc(sizeof(float) * BB * HD);
    float* q   = (float*)alloc(sizeof(float) * BB * HD);
    _Float16* enc_outh = (_Float16*)alloc(2 * (size_t)BB * SS * 2 * HE);
    _Float16* h16_f = (_Float16*)alloc(2 * BB * HE);
    _Float16* h16_b = (_Float16*)alloc(2 * BB * HE);
    _Float16* h0h = (_Float16*)alloc(2 * BB * HD);
    _Float16* h1h = (_Float16*)alloc(2 * BB * HD);
    _Float16* h2h = (_Float16*)alloc(2 * BB * HD);
    _Float16* inp16 = (_Float16*)alloc(2 * BB * (ED + HD));
    _Float16* whh16_f = (_Float16*)alloc(2 * 3 * HE * HE);
    _Float16* whh16_b = (_Float16*)alloc(2 * 3 * HE * HE);
    _Float16* w1_16 = (_Float16*)alloc(2 * HD * HD);
    _Float16* w2_16 = (_Float16*)alloc(2 * HD * HD);
    _Float16* wih16[3], *whh16[3];
    int inKs[3] = {ED + HD, HD, HD};
    for (int l = 0; l < 3; ++l) {
        wih16[l] = (_Float16*)alloc(2 * 3 * HD * inKs[l]);
        whh16[l] = (_Float16*)alloc(2 * 3 * HD * HD);
    }
    _Float16* owp16 = (_Float16*)alloc(2 * 112 * HD);

    auto cvt = [&](const float* src, _Float16* dst, int dr, int dc, int sr, int sc, int co) {
        int n = dr * dc;
        cvt_kernel<<<(n + 255) / 256, 256, 0, stream>>>(src, dst, dr, dc, sr, sc, co);
    };
    // ---- weight conversions (once per call) ----
    cvt(eWhh_f, whh16_f, 3 * HE, HE, 3 * HE, HE, 0);
    cvt(eWhh_b, whh16_b, 3 * HE, HE, 3 * HE, HE, 0);
    cvt(attn_W, w1_16, HD, HD, HD, 2 * HD, 0);
    cvt(attn_W, w2_16, HD, HD, HD, 2 * HD, HD);
    for (int l = 0; l < 3; ++l) {
        cvt(dWih[l], wih16[l], 3 * HD, inKs[l], 3 * HD, inKs[l], 0);
        cvt(dWhh[l], whh16[l], 3 * HD, HD, 3 * HD, HD, 0);
    }
    cvt(out_W, owp16, 112, HD, VC, HD, 0);

    // ---- init encoder hidden states (f32 + f16 mirror) ----
    zero_pair_kernel<<<(BB * HE + 255) / 256, 256, 0, stream>>>(h_f, h16_f, BB * HE);
    zero_pair_kernel<<<(BB * HE + 255) / 256, 256, 0, stream>>>(h_b, h16_b, BB * HE);

    // ---- bidirectional encoder, 120 fused steps ----
    for (int t = 0; t < SS; ++t)
        enc_step_kernel<<<dim3(8, 2), 512, 0, stream>>>(
            tgts, emb_enc,
            eWih_f, ebih_f, ebhh_f, whh16_f, h_f, h16_f,
            eWih_b, ebih_b, ebhh_b, whh16_b, h_b, h16_b,
            enc_out, enc_outh, t);

    // ---- attention precompute: pre_enc = enc_out @ W2^T + attn_b ----
    gemm_kernel<<<dim3(BB * SS / 16, HD / 128), 256, 0, stream>>>(enc_outh, w2_16, attn_b,
                                                                  pre_enc, HD, HD);
    // ---- decoder hidden init: h0=[h_f,h_b], h1=h2=0 ----
    h0_init_kernel<<<(BB * HD + 255) / 256, 256, 0, stream>>>(h_f, h_b, h0, h0h);
    zero_pair_kernel<<<(BB * HD + 255) / 256, 256, 0, stream>>>(h1, h1h, BB * HD);
    zero_pair_kernel<<<(BB * HD + 255) / 256, 256, 0, stream>>>(h2, h2h, BB * HD);

    // ---- decoder, 120 steps ----
    for (int t = 0; t < TT; ++t) {
        gemm_kernel<<<dim3(BB / 16, HD / 128), 256, 0, stream>>>(h0h, w1_16, nullptr, q, HD, HD);
        attn_ctx_kernel<<<BB, 128, 0, stream>>>(q, pre_enc, attn_v, enc_out, emb_dec, cmps, inp16, t);
        dec_gru_kernel<<<8, 512, 0, stream>>>(inp16, ED + HD, wih16[0], whh16[0], dbih[0], dbhh[0], h0, h0h);
        dec_gru_kernel<<<8, 512, 0, stream>>>(h0h, HD, wih16[1], whh16[1], dbih[1], dbhh[1], h1, h1h);
        dec_gru_kernel<<<8, 512, 0, stream>>>(h1h, HD, wih16[2], whh16[2], dbih[2], dbhh[2], h2, h2h);
        logits_kernel<<<8, 256, 0, stream>>>(h2h, owp16, out_b, out, t);
    }
    (void)in_sizes; (void)n_in; (void)out_size; (void)ws_size;
}
